// Net_56547539419822
// MI455X (gfx1250) — matrix operand order repaired and checked by reading, not compiled, not run on hardware
//
#include <hip/hip_runtime.h>
#include <math.h>

typedef __attribute__((ext_vector_type(16))) _Float16 v16h;
typedef __attribute__((ext_vector_type(8)))  float    v8f;

// ---------------------------------------------------------------- utilities

__device__ __forceinline__ void atomAddF(float* p, float v) {
  unsafeAtomicAdd(p, v);   // hardware global_atomic_add_f32
}

__global__ void k_zero(float* __restrict__ p, long n) {
  long i = (long)blockIdx.x * blockDim.x + threadIdx.x;
  long stride = (long)gridDim.x * blockDim.x;
  for (; i < n; i += stride) p[i] = 0.f;
}

__global__ void k_degree(const int* __restrict__ ei, float* __restrict__ cnt, int nE) {
  int e = blockIdx.x * blockDim.x + threadIdx.x;
  if (e < nE) atomAddF(&cnt[ei[nE + e]], 1.0f);
}

__global__ void k_inv(const float* __restrict__ cnt, float* __restrict__ inv, int n) {
  int i = blockIdx.x * blockDim.x + threadIdx.x;
  if (i < n) inv[i] = 1.0f / fmaxf(cnt[i], 1.0f);
}

// One wave (32 lanes) per edge; lanes stride over float4 chunks of the row.
template <int F>
__global__ void k_scatter(const float* __restrict__ h, const int* __restrict__ ei,
                          float* __restrict__ s, int nE) {
  constexpr int C = F / 4;                     // float4 chunks per row
  int wid  = (int)(((long)blockIdx.x * blockDim.x + threadIdx.x) >> 5);
  int lane = threadIdx.x & 31;
  if (wid >= nE) return;
  long src = ei[wid];
  long dst = ei[nE + wid];
  const float4* hp = (const float4*)(h + src * F);
  float*        sp = s + dst * F;
  for (int c = lane; c < C; c += 32) {
    float4 v = hp[c];
    atomAddF(&sp[c * 4 + 0], v.x);
    atomAddF(&sp[c * 4 + 1], v.y);
    atomAddF(&sp[c * 4 + 2], v.z);
    atomAddF(&sp[c * 4 + 3], v.w);
  }
}

// ------------------------------------------------------- fused SAGE GEMM
// out[n, FOUT] = (s / cnt) @ Wl + x @ Wr + bias
// Two K passes (agg then self) into two accumulator sets; the per-row mean
// divide (inv) is applied at the epilogue: out = acc0*inv[row] + acc1.
// B is staged once per block into LDS, column-major f16, zero-padded for
// k in [K1, K1P) and col in [FOUT, NT*16) -- A may load clamped garbage in
// those slots without affecting the result (it multiplies zero B).
template <int K1, int FOUT, int NT>
__global__ __launch_bounds__(256)
void k_sage_gemm(const float* __restrict__ s, const float* __restrict__ inv,
                 const float* __restrict__ xin,
                 const float* __restrict__ Wl, const float* __restrict__ Wr,
                 const float* __restrict__ bias,
                 float* __restrict__ out, int n) {
  constexpr int K1P     = (K1 + 31) & ~31;   // per-section padded K
  constexpr int KT      = K1P / 32;
  constexpr int NCOL    = NT * 16;
  constexpr int BSTRIDE = 2 * K1P;           // halfs per column (2 sections)

  __shared__ _Float16 Bsh[NCOL * BSTRIDE];

  // ---- cooperative B stage: Bsh[col][sec*K1P + k] = Wsec[k][col] ----
  for (int idx = threadIdx.x; idx < NCOL * BSTRIDE; idx += blockDim.x) {
    int col = idx / BSTRIDE;
    int kk  = idx % BSTRIDE;
    int sec = (kk >= K1P) ? 1 : 0;
    int k   = kk - sec * K1P;
    float v = 0.f;
    if (col < FOUT && k < K1)
      v = sec ? Wr[(long)k * FOUT + col] : Wl[(long)k * FOUT + col];
    Bsh[idx] = (_Float16)v;
  }
  __syncthreads();

  int lane = threadIdx.x & 31;
  int half = lane >> 4;
  int c15  = lane & 15;

  long rowBase = ((long)blockIdx.x * (blockDim.x >> 5) + (threadIdx.x >> 5)) * 16;

  v8f acc0[NT];   // aggregated-neighbor partial (scaled by inv at epilogue)
  v8f acc1[NT];   // self partial + bias
#pragma unroll
  for (int nt = 0; nt < NT; ++nt) {
    int col  = nt * 16 + c15;
    float bv = (col < FOUT) ? bias[col] : 0.f;
#pragma unroll
    for (int r = 0; r < 8; ++r) { acc0[nt][r] = 0.f; acc1[nt][r] = bv; }
  }

  long m  = rowBase + c15;
  long mc = (m < n) ? m : (long)(n - 1);     // clamp; garbage rows never stored

  // Base LDS offset for this lane's B fragments (element e <-> k = 16*half + e).
  const _Float16* bBase = &Bsh[c15 * BSTRIDE + half * 16];

#pragma unroll
  for (int sec = 0; sec < 2; ++sec) {
    const float* rowPtr = (sec ? xin : s) + mc * K1;
    v8f* accp = sec ? acc1 : acc0;

#pragma unroll
    for (int kt = 0; kt < KT; ++kt) {
      // A fragment: elements 0..7  <-> k = kt*32 + 16*half + e
      //             elements 8..15 <-> k = kt*32 + 16*half + 8 + e
      int o0 = kt * 32 + half * 16;
      int o1 = o0 + 8;
      if (kt == KT - 1 && (K1 % 32) != 0) {  // clamp only on boundary tile
        o0 = (o0 < K1 - 8) ? o0 : (K1 - 8);
        o1 = (o1 < K1 - 8) ? o1 : (K1 - 8);
      }
      float4 r0 = *(const float4*)(rowPtr + o0);
      float4 r1 = *(const float4*)(rowPtr + o0 + 4);
      float4 r2 = *(const float4*)(rowPtr + o1);
      float4 r3 = *(const float4*)(rowPtr + o1 + 4);

      v16h a;
      a[0]  = (_Float16)r0.x; a[1]  = (_Float16)r0.y;
      a[2]  = (_Float16)r0.z; a[3]  = (_Float16)r0.w;
      a[4]  = (_Float16)r1.x; a[5]  = (_Float16)r1.y;
      a[6]  = (_Float16)r1.z; a[7]  = (_Float16)r1.w;
      a[8]  = (_Float16)r2.x; a[9]  = (_Float16)r2.y;
      a[10] = (_Float16)r2.z; a[11] = (_Float16)r2.w;
      a[12] = (_Float16)r3.x; a[13] = (_Float16)r3.y;
      a[14] = (_Float16)r3.z; a[15] = (_Float16)r3.w;

      // Batch-load all NT B fragments (one LDS clause), then issue the WMMAs.
      v16h bfr[NT];
#pragma unroll
      for (int nt = 0; nt < NT; ++nt)
        bfr[nt] = *(const v16h*)(bBase + nt * 16 * BSTRIDE + sec * K1P + kt * 32);

#pragma unroll
      for (int nt = 0; nt < NT; ++nt)
        accp[nt] = __builtin_amdgcn_wmma_f32_16x16x32_f16(
            false, a, false, bfr[nt], (short)0, accp[nt], false, false);
    }
  }

  // Epilogue: per-row mean divide on the agg partial, then store.
  float invRow[8];
#pragma unroll
  for (int r = 0; r < 8; ++r) {
    long row = rowBase + half * 8 + r;
    invRow[r] = inv[(row < n) ? row : (n - 1)];
  }
#pragma unroll
  for (int nt = 0; nt < NT; ++nt) {
    int col = nt * 16 + c15;
#pragma unroll
    for (int r = 0; r < 8; ++r) {
      long row = rowBase + half * 8 + r;
      if (row < n && col < FOUT)
        out[row * FOUT + col] = acc0[nt][r] * invRow[r] + acc1[nt][r];
    }
  }
}

// ------------------------------------------------------------ batch norm

template <int F>
__global__ __launch_bounds__(256)
void k_bn_stats(const float* __restrict__ h, float* __restrict__ stats, int n) {
  __shared__ float ssum[256];
  __shared__ float ssq[256];
  const int G = 256 / F;
  int tid = threadIdx.x;
  int f   = tid & (F - 1);
  int g   = tid / F;

  float sum = 0.f, sq = 0.f;
  for (long r = (long)blockIdx.x * G + g; r < n; r += (long)gridDim.x * G) {
    float v = h[r * F + f];
    sum += v; sq += v * v;
  }
  ssum[tid] = sum; ssq[tid] = sq;
  __syncthreads();
  for (int off = 128; off >= F; off >>= 1) {
    if (tid < off) { ssum[tid] += ssum[tid + off]; ssq[tid] += ssq[tid + off]; }
    __syncthreads();
  }
  if (tid < F) {
    atomAddF(&stats[f],     ssum[tid]);
    atomAddF(&stats[F + f], ssq[tid]);
  }
}

template <int F>
__global__ void k_bn_final(const float* __restrict__ stats,
                           const float* __restrict__ gamma,
                           const float* __restrict__ beta,
                           float* __restrict__ bnA, float* __restrict__ bnB, int n) {
  int f = threadIdx.x;
  if (f < F) {
    float mean = stats[f] / (float)n;
    float var  = stats[F + f] / (float)n - mean * mean;  // biased variance
    float A    = gamma[f] * rsqrtf(var + 1e-5f);
    bnA[f] = A;
    bnB[f] = beta[f] - mean * A;
  }
}

template <int F>
__global__ void k_bn_relu(float* __restrict__ h, const float* __restrict__ bnA,
                          const float* __restrict__ bnB, long total) {
  long i = (long)blockIdx.x * blockDim.x + threadIdx.x;
  long stride = (long)gridDim.x * blockDim.x;
  for (; i < total; i += stride) {
    int f   = (int)i & (F - 1);          // F is a power of two
    float v = h[i] * bnA[f] + bnB[f];
    h[i] = v > 0.f ? v : 0.f;
  }
}

// ---------------------------------------------------------- log-softmax (17)

__global__ void k_logsoftmax17(float* __restrict__ out, int n) {
  int r = blockIdx.x * blockDim.x + threadIdx.x;
  if (r >= n) return;
  float v[17];
  float mx = -INFINITY;
#pragma unroll
  for (int c = 0; c < 17; ++c) { v[c] = out[(long)r * 17 + c]; mx = fmaxf(mx, v[c]); }
  float ssum = 0.f;
#pragma unroll
  for (int c = 0; c < 17; ++c) ssum += __expf(v[c] - mx);
  float lg = __logf(ssum);
#pragma unroll
  for (int c = 0; c < 17; ++c) out[(long)r * 17 + c] = v[c] - mx - lg;
}

// ---------------------------------------------------------------- launcher

static inline int cdiv(long a, long b) { return (int)((a + b - 1) / b); }

extern "C" void kernel_launch(void* const* d_in, const int* in_sizes, int n_in,
                              void* d_out, int out_size, void* d_ws, size_t ws_size,
                              hipStream_t stream) {
  const float* x   = (const float*)d_in[0];
  const int*   ei  = (const int*)d_in[1];
  const float* W1l = (const float*)d_in[2];
  const float* b1  = (const float*)d_in[3];
  const float* W1r = (const float*)d_in[4];
  const float* g1  = (const float*)d_in[5];
  const float* be1 = (const float*)d_in[6];
  const float* W2l = (const float*)d_in[7];
  const float* b2  = (const float*)d_in[8];
  const float* W2r = (const float*)d_in[9];
  const float* g2  = (const float*)d_in[10];
  const float* be2 = (const float*)d_in[11];
  const float* W3l = (const float*)d_in[12];
  const float* b3  = (const float*)d_in[13];
  const float* W3r = (const float*)d_in[14];

  const int IN = 200;
  const int n  = in_sizes[0] / IN;      // 100000
  const int E  = in_sizes[1] / 2;       // 1600000

  // Workspace layout (floats).
  float* ws    = (float*)d_ws;
  float* cnt   = ws;                         // n
  float* inv   = cnt + n;                    // n
  float* stats = inv + n;                    // 128
  float* bnA   = stats + 128;                // 64
  float* bnB   = bnA + 64;                   // 64
  float* big   = bnB + 64;                   // n*200 (scatter scratch, reused)
  float* h1    = big + (long)n * IN;         // n*64
  float* h2    = h1 + (long)n * 64;          // n*32

  float* logits = (float*)d_out;             // n*17

  const int TB = 256;
  int gN    = cdiv(n, TB);
  int gE    = cdiv(E, TB);
  int gEdge = cdiv((long)E * 32, TB);        // one wave per edge
  int gGemm = cdiv(n, 16 * (TB / 32));       // 16 rows per wave, 8 waves/block

  // Degree and inverse-degree (shared across layers).
  k_zero<<<gN, TB, 0, stream>>>(cnt, n);
  k_degree<<<gE, TB, 0, stream>>>(ei, cnt, E);
  k_inv<<<gN, TB, 0, stream>>>(cnt, inv, n);

  // ---------------- Layer 1: 200 -> 64 ----------------
  k_zero<<<cdiv((long)n * IN, TB * 4), TB, 0, stream>>>(big, (long)n * IN);
  k_scatter<200><<<gEdge, TB, 0, stream>>>(x, ei, big, E);
  k_sage_gemm<200, 64, 4><<<gGemm, TB, 0, stream>>>(big, inv, x, W1l, W1r, b1, h1, n);
  k_zero<<<1, 128, 0, stream>>>(stats, 128);
  k_bn_stats<64><<<512, TB, 0, stream>>>(h1, stats, n);
  k_bn_final<64><<<1, 64, 0, stream>>>(stats, g1, be1, bnA, bnB, n);
  k_bn_relu<64><<<cdiv((long)n * 64, TB * 4), TB, 0, stream>>>(h1, bnA, bnB, (long)n * 64);

  // ---------------- Layer 2: 64 -> 32 ----------------
  k_zero<<<cdiv((long)n * 64, TB * 4), TB, 0, stream>>>(big, (long)n * 64);
  k_scatter<64><<<gEdge, TB, 0, stream>>>(h1, ei, big, E);
  k_sage_gemm<64, 32, 2><<<gGemm, TB, 0, stream>>>(big, inv, h1, W2l, W2r, b2, h2, n);
  k_zero<<<1, 128, 0, stream>>>(stats, 128);
  k_bn_stats<32><<<512, TB, 0, stream>>>(h2, stats, n);
  k_bn_final<32><<<1, 32, 0, stream>>>(stats, g2, be2, bnA, bnB, n);
  k_bn_relu<32><<<cdiv((long)n * 32, TB * 4), TB, 0, stream>>>(h2, bnA, bnB, (long)n * 32);

  // ---------------- Layer 3: 32 -> 17 + log-softmax ----------------
  k_zero<<<cdiv((long)n * 32, TB * 4), TB, 0, stream>>>(big, (long)n * 32);
  k_scatter<32><<<gEdge, TB, 0, stream>>>(h2, ei, big, E);
  k_sage_gemm<32, 17, 2><<<gGemm, TB, 0, stream>>>(big, inv, h2, W3l, W3r, b3, logits, n);
  k_logsoftmax17<<<gN, TB, 0, stream>>>(logits, n);
}
